// MambaV7_39556648796759
// MI455X (gfx1250) — matrix operand order repaired and checked by reading, not compiled, not run on hardware
//
#include <hip/hip_runtime.h>
#include <hip/hip_bf16.h>

typedef _Float16 half_t;
typedef __attribute__((ext_vector_type(16))) _Float16      v16h;
typedef __attribute__((ext_vector_type(8)))  float         v8f;
typedef __attribute__((ext_vector_type(4)))  unsigned int  v4u;

#define BATCH   64
#define SEQ     1024
#define TOKENS  (BATCH * SEQ)      // 65536
#define D_MODEL 128
#define D_INNER 256
#define D_STATE 16
#define D_CONV  4
#define DT_RANK 8
#define XDBL_PAD 48                // dt(8) + B(16) + C(16) padded to 48 for WMMA N-tiling

// ---------------------------------------------------------------------------
// Encoder: h = LN(relu(x @ enc_w + enc_b)) -> f16 for WMMA.  1 block / token.
// ---------------------------------------------------------------------------
__global__ void encoder_kernel(const float* __restrict__ x,
                               const float* __restrict__ enc_w,
                               const float* __restrict__ enc_b,
                               const float* __restrict__ enc_g,
                               const float* __restrict__ enc_beta,
                               half_t* __restrict__ h16) {
    const int t = blockIdx.x;      // token
    const int d = threadIdx.x;     // 0..127
    __shared__ float xr[16];
    __shared__ float red[D_MODEL];

    if (d < 16) xr[d] = x[(size_t)t * 16 + d];
    __syncthreads();

    float acc = enc_b[d];
#pragma unroll
    for (int i = 0; i < 16; ++i) acc += xr[i] * enc_w[i * D_MODEL + d];
    acc = fmaxf(acc, 0.0f);

    red[d] = acc; __syncthreads();
#pragma unroll
    for (int s = 64; s > 0; s >>= 1) { if (d < s) red[d] += red[d + s]; __syncthreads(); }
    const float mu = red[0] * (1.0f / D_MODEL);
    __syncthreads();
    const float cdev = acc - mu;
    red[d] = cdev * cdev; __syncthreads();
#pragma unroll
    for (int s = 64; s > 0; s >>= 1) { if (d < s) red[d] += red[d + s]; __syncthreads(); }
    const float var = red[0] * (1.0f / D_MODEL);
    const float o = cdev * rsqrtf(var + 1e-5f) * enc_g[d] + enc_beta[d];
    h16[(size_t)t * D_MODEL + d] = (half_t)o;
}

// ---------------------------------------------------------------------------
// f32 -> f16 convert (activations)
// ---------------------------------------------------------------------------
__global__ void f32_to_f16(const float* __restrict__ in, half_t* __restrict__ out, int n) {
    int i = blockIdx.x * blockDim.x + threadIdx.x;
    if (i < n) out[i] = (half_t)in[i];
}

// ---------------------------------------------------------------------------
// Pack a row-major f32 weight (K x cols) into WMMA-B fragment order (f16):
//   out[ ((kc*tilesN + tn)*32 + lane)*16 + e ] = B[kc*32 + 16*(e>>3) + 8*(lane>>4) + (e&7)][tn*16 + (lane&15)]
// Zero-pads columns up to Npad.  Each lane's fragment becomes 32 contiguous bytes.
// ---------------------------------------------------------------------------
__global__ void pack_b_wmma(const float* __restrict__ in, half_t* __restrict__ out,
                            int K, int cols, int Npad) {
    const int idx = blockIdx.x * blockDim.x + threadIdx.x;
    const int tilesN = Npad >> 4;
    if (idx >= K * Npad) return;
    const int e    = idx & 15;
    const int lane = (idx >> 4) & 31;
    const int rest = idx >> 9;               // kc*tilesN + tn
    const int tn   = rest % tilesN;
    const int kc   = rest / tilesN;
    const int ksrc = kc * 32 + ((e >> 3) << 4) + ((lane >> 4) << 3) + (e & 7);
    const int col  = tn * 16 + (lane & 15);
    const float v  = (col < cols) ? in[(size_t)ksrc * cols + col] : 0.0f;
    out[idx] = (half_t)v;
}

// ---------------------------------------------------------------------------
// WMMA GEMM: C[M,N](f32) = A[M,K](f16, row-major) x Bsw (pre-swizzled f16).
// Wave computes a 16 x (16*NT) strip; K fully unrolled; b128 fragment loads.
// ---------------------------------------------------------------------------
template<int K, int NT>
__global__ __launch_bounds__(256)
void wmma_gemm_f16(const half_t* __restrict__ A, const half_t* __restrict__ Bsw,
                   float* __restrict__ C, int M, int N) {
    constexpr int KC = K / 32;
    const int lane    = threadIdx.x & 31;
    const int wave    = threadIdx.x >> 5;
    const int tilesN  = N >> 4;
    const int groupsN = tilesN / NT;
    const int wid     = blockIdx.x * (blockDim.x >> 5) + wave;
    if (wid >= (M >> 4) * groupsN) return;     // wave-uniform: EXEC stays all-1s

    const int tm      = wid / groupsN;
    const int tn0     = (wid % groupsN) * NT;
    const int m       = lane & 15;
    const int halfsel = lane >> 4;

    const v4u* Arow4 = (const v4u*)(A + (size_t)(tm * 16 + m) * K);   // 8 halfs / v4u
    const v4u* B4    = (const v4u*)Bsw;

    union Frag { v4u q[2]; v16h v; };
    v8f acc[NT];
    const v8f vzero = {};
#pragma unroll
    for (int j = 0; j < NT; ++j) acc[j] = vzero;

#pragma unroll
    for (int kc = 0; kc < KC; ++kc) {
        if (kc + 1 < KC) {                     // gfx1250 global_prefetch_b8
            __builtin_prefetch(&Arow4[(kc + 1) * 4 + halfsel], 0, 0);
            __builtin_prefetch(&B4[((size_t)((kc + 1) * tilesN + tn0) * 32 + lane) * 2], 0, 0);
        }
        Frag a;
        a.q[0] = Arow4[kc * 4 + halfsel];          // K = kc*32 + 8*halfsel .. +7
        a.q[1] = Arow4[kc * 4 + 2 + halfsel];      // K = kc*32 + 16 + 8*halfsel .. +7
#pragma unroll
        for (int j = 0; j < NT; ++j) {
            Frag b;
            const size_t bi = ((size_t)(kc * tilesN + tn0 + j) * 32 + lane) * 2;
            b.q[0] = B4[bi];
            b.q[1] = B4[bi + 1];
            acc[j] = __builtin_amdgcn_wmma_f32_16x16x32_f16(
                /*neg_a=*/false, a.v, /*neg_b=*/false, b.v,
                /*c_mod=*/(short)0, acc[j], /*reuse_a=*/false, /*reuse_b=*/false);
        }
    }
    // C/D layout: VGPR r -> M = r + 8*halfsel, N = lane&15
    float* Cb = C + (size_t)(tm * 16) * N + m;
#pragma unroll
    for (int j = 0; j < NT; ++j)
#pragma unroll
        for (int r = 0; r < 8; ++r)
            Cb[(size_t)(r + 8 * halfsel) * N + (tn0 + j) * 16] = acc[j][r];
}

// ---------------------------------------------------------------------------
// Causal depthwise conv1d (width 4) + SiLU over the xx half of xz -> f16
// ---------------------------------------------------------------------------
__global__ void conv_silu_kernel(const float* __restrict__ xz,
                                 const float* __restrict__ conv_w,
                                 const float* __restrict__ conv_b,
                                 half_t* __restrict__ xx16) {
    const int idx = blockIdx.x * blockDim.x + threadIdx.x;   // over TOKENS*D_INNER
    if (idx >= TOKENS * D_INNER) return;
    const int d = idx & (D_INNER - 1);
    const int t = idx >> 8;               // token index (D_INNER == 256)
    const int l = t & (SEQ - 1);

    float acc = conv_b[d];
#pragma unroll
    for (int j = 0; j < D_CONV; ++j) {
        const int ll = l - (D_CONV - 1) + j;
        if (ll >= 0)
            acc += conv_w[d * D_CONV + j] * xz[(size_t)(t - (D_CONV - 1) + j) * (2 * D_INNER) + d];
    }
    const float s = acc / (1.0f + __expf(-acc));   // silu
    xx16[idx] = (half_t)s;
}

// ---------------------------------------------------------------------------
// Selective scan, fused: dt = softplus(xdbl[:,:8] @ dt_w + dt_b),
// h = exp(dt*A)h + (dt*xx) B ; y = <h,C> ; out = (y + xx*D) * silu(z) -> f16
// One block per batch element (256 threads = d_inner), state in registers.
// ---------------------------------------------------------------------------
__global__ __launch_bounds__(256)
void scan_kernel(const float* __restrict__ xdbl,     // TOKENS x XDBL_PAD
                 const half_t* __restrict__ xx16,    // TOKENS x D_INNER
                 const float* __restrict__ xz,       // TOKENS x 2*D_INNER (z in cols 256..511)
                 const float* __restrict__ dt_w,     // DT_RANK x D_INNER
                 const float* __restrict__ dt_b,     // D_INNER
                 const float* __restrict__ A_log,    // D_INNER x D_STATE
                 const float* __restrict__ Dp,       // D_INNER
                 half_t* __restrict__ y16) {         // TOKENS x D_INNER
    const int b = blockIdx.x;       // 0..63
    const int d = threadIdx.x;      // 0..255
    __shared__ float row[XDBL_PAD];

    float A[D_STATE];
#pragma unroll
    for (int s = 0; s < D_STATE; ++s) A[s] = -__expf(A_log[d * D_STATE + s]);
    float wdt[DT_RANK];
#pragma unroll
    for (int r = 0; r < DT_RANK; ++r) wdt[r] = dt_w[r * D_INNER + d];
    const float bdt = dt_b[d];
    const float Dd  = Dp[d];

    float h[D_STATE];
#pragma unroll
    for (int s = 0; s < D_STATE; ++s) h[s] = 0.0f;

    for (int l = 0; l < SEQ; ++l) {
        const size_t t = (size_t)b * SEQ + l;
        __syncthreads();
        if (d < XDBL_PAD) row[d] = xdbl[t * XDBL_PAD + d];
        __syncthreads();

        float dtp = bdt;
#pragma unroll
        for (int r = 0; r < DT_RANK; ++r) dtp += row[r] * wdt[r];
        dtp = (dtp > 20.0f) ? dtp : log1pf(__expf(dtp));      // softplus

        const float xv = (float)xx16[t * D_INNER + d];
        const float du = dtp * xv;
        float y = 0.0f;
#pragma unroll
        for (int s = 0; s < D_STATE; ++s) {
            const float dA = __expf(dtp * A[s]);
            h[s] = dA * h[s] + du * row[DT_RANK + s];          // B_t
            y += h[s] * row[DT_RANK + D_STATE + s];            // C_t
        }
        const float zv  = xz[t * (2 * D_INNER) + D_INNER + d];
        const float out = (y + xv * Dd) * (zv / (1.0f + __expf(-zv)));
        y16[t * D_INNER + d] = (half_t)out;
    }
}

// ---------------------------------------------------------------------------
// Head: LN(last token) -> relu(@h1) @h2 + skip.  1 block / batch element.
// ---------------------------------------------------------------------------
__global__ void head_kernel(const float* __restrict__ h32, const float* __restrict__ x,
                            const float* __restrict__ norm_g, const float* __restrict__ norm_b,
                            const float* __restrict__ h1_w, const float* __restrict__ h1_b,
                            const float* __restrict__ h2_w, const float* __restrict__ h2_b,
                            const float* __restrict__ skip_w, const float* __restrict__ skip_b,
                            float* __restrict__ out) {
    const int b = blockIdx.x;      // 0..63
    const int d = threadIdx.x;     // 0..127
    __shared__ float red[D_MODEL];
    __shared__ float lnv[D_MODEL];
    __shared__ float hid[32];
    const size_t t = (size_t)b * SEQ + (SEQ - 1);

    const float v = h32[t * D_MODEL + d];
    red[d] = v; __syncthreads();
#pragma unroll
    for (int s = 64; s > 0; s >>= 1) { if (d < s) red[d] += red[d + s]; __syncthreads(); }
    const float mu = red[0] * (1.0f / D_MODEL);
    __syncthreads();
    const float cdev = v - mu;
    red[d] = cdev * cdev; __syncthreads();
#pragma unroll
    for (int s = 64; s > 0; s >>= 1) { if (d < s) red[d] += red[d + s]; __syncthreads(); }
    const float var = red[0] * (1.0f / D_MODEL);
    lnv[d] = cdev * rsqrtf(var + 1e-5f) * norm_g[d] + norm_b[d];
    __syncthreads();

    if (d < 32) {
        float a = h1_b[d];
        for (int i = 0; i < D_MODEL; ++i) a += lnv[i] * h1_w[i * 32 + d];
        hid[d] = fmaxf(a, 0.0f);
    }
    __syncthreads();
    if (d == 0) {
        float a = h2_b[0];
#pragma unroll
        for (int j = 0; j < 32; ++j) a += hid[j] * h2_w[j];
        float sk = skip_b[0];
#pragma unroll
        for (int i = 0; i < 16; ++i) sk += x[t * 16 + i] * skip_w[i];
        out[b] = a + sk;
    }
}

// ---------------------------------------------------------------------------
// Host orchestration
// ---------------------------------------------------------------------------
static inline int gemm_blocks(int M, int N, int NT) {
    int waves = (M >> 4) * ((N >> 4) / NT);
    return (waves + 7) / 8;                    // 8 waves per 256-thread block
}

extern "C" void kernel_launch(void* const* d_in, const int* in_sizes, int n_in,
                              void* d_out, int out_size, void* d_ws, size_t ws_size,
                              hipStream_t stream) {
    const float* x        = (const float*)d_in[0];
    const float* enc_w    = (const float*)d_in[1];
    const float* enc_b    = (const float*)d_in[2];
    const float* enc_g    = (const float*)d_in[3];
    const float* enc_beta = (const float*)d_in[4];
    const float* norm_g   = (const float*)d_in[23];
    const float* norm_b   = (const float*)d_in[24];
    const float* h1_w     = (const float*)d_in[25];
    const float* h1_b     = (const float*)d_in[26];
    const float* h2_w     = (const float*)d_in[27];
    const float* h2_b     = (const float*)d_in[28];
    const float* skip_w   = (const float*)d_in[29];
    const float* skip_b   = (const float*)d_in[30];

    // workspace layout (bytes); all offsets 256B-aligned
    char* ws = (char*)d_ws;
    const size_t T = TOKENS;
    size_t off = 0;
    half_t* h16   = (half_t*)(ws + off); off += T * D_MODEL * sizeof(half_t);        // 16.8 MB
    float*  xz    = (float* )(ws + off); off += T * 2 * D_INNER * sizeof(float);     // 134 MB
    half_t* xx16  = (half_t*)(ws + off); off += T * D_INNER * sizeof(half_t);        // 33.6 MB
    float*  xdbl  = (float* )(ws + off); off += T * XDBL_PAD * sizeof(float);        // 12.6 MB
    half_t* y16   = (half_t*)(ws + off); off += T * D_INNER * sizeof(half_t);        // 33.6 MB
    float*  h32   = (float* )(ws + off); off += T * D_MODEL * sizeof(float);         // 33.6 MB
    half_t* w16in = (half_t*)(ws + off); off += (size_t)D_MODEL * 2 * D_INNER * sizeof(half_t);
    half_t* w16xp = (half_t*)(ws + off); off += (size_t)D_INNER * XDBL_PAD * sizeof(half_t);
    half_t* w16o  = (half_t*)(ws + off); off += (size_t)D_INNER * D_MODEL * sizeof(half_t);
    (void)ws_size; (void)in_sizes; (void)n_in; (void)out_size;

    // 1) encoder -> h16
    encoder_kernel<<<TOKENS, D_MODEL, 0, stream>>>(x, enc_w, enc_b, enc_g, enc_beta, h16);

    // 2) two mamba blocks
    for (int p = 0; p < 2; ++p) {
        const int wb = 5 + p * 9;
        const float* in_w    = (const float*)d_in[wb + 0];
        const float* conv_w  = (const float*)d_in[wb + 1];
        const float* conv_b  = (const float*)d_in[wb + 2];
        const float* xproj_w = (const float*)d_in[wb + 3];
        const float* dt_w    = (const float*)d_in[wb + 4];
        const float* dt_b    = (const float*)d_in[wb + 5];
        const float* A_log   = (const float*)d_in[wb + 6];
        const float* Dp      = (const float*)d_in[wb + 7];
        const float* out_w   = (const float*)d_in[wb + 8];

        // in_proj GEMM: (T x 128) @ (128 x 512) -> xz
        {
            int n = D_MODEL * 2 * D_INNER;
            pack_b_wmma<<<(n + 255) / 256, 256, 0, stream>>>(
                in_w, w16in, D_MODEL, 2 * D_INNER, 2 * D_INNER);
            wmma_gemm_f16<D_MODEL, 4><<<gemm_blocks(TOKENS, 2 * D_INNER, 4), 256, 0, stream>>>(
                h16, w16in, xz, TOKENS, 2 * D_INNER);
        }
        // depthwise conv + silu -> xx16
        conv_silu_kernel<<<(TOKENS * D_INNER) / 256, 256, 0, stream>>>(xz, conv_w, conv_b, xx16);

        // x_proj GEMM: (T x 256) @ (256 x 48pad) -> xdbl
        {
            int n = D_INNER * XDBL_PAD;
            pack_b_wmma<<<(n + 255) / 256, 256, 0, stream>>>(
                xproj_w, w16xp, D_INNER, DT_RANK + 2 * D_STATE, XDBL_PAD);
            wmma_gemm_f16<D_INNER, 3><<<gemm_blocks(TOKENS, XDBL_PAD, 3), 256, 0, stream>>>(
                xx16, w16xp, xdbl, TOKENS, XDBL_PAD);
        }
        // fused selective scan (+dt proj, +D residual, +silu(z) gate) -> y16
        scan_kernel<<<BATCH, D_INNER, 0, stream>>>(
            xdbl, xx16, xz, dt_w, dt_b, A_log, Dp, y16);

        // out_proj GEMM: (T x 256) @ (256 x 128) -> h32
        {
            int n = D_INNER * D_MODEL;
            pack_b_wmma<<<(n + 255) / 256, 256, 0, stream>>>(
                out_w, w16o, D_INNER, D_MODEL, D_MODEL);
            wmma_gemm_f16<D_INNER, 4><<<gemm_blocks(TOKENS, D_MODEL, 4), 256, 0, stream>>>(
                y16, w16o, h32, TOKENS, D_MODEL);
        }
        // next block consumes f16 activations
        {
            int n = TOKENS * D_MODEL;
            f32_to_f16<<<(n + 255) / 256, 256, 0, stream>>>(h32, h16, n);
        }
    }

    // 3) head: LN(last) -> MLP + skip
    head_kernel<<<BATCH, D_MODEL, 0, stream>>>(
        h32, x, norm_g, norm_b, h1_w, h1_b, h2_w, h2_b, skip_w, skip_b, (float*)d_out);
}
